// DWConvPWConv_15126874817124
// MI455X (gfx1250) — 8-bit matrix operand order repaired and checked by reading, compiled, not run on hardware
//
#include <hip/hip_runtime.h>
#include <cstdint>
#include <cstddef>

typedef __attribute__((ext_vector_type(8))) int v8i;

#define C_IN   256
#define C_OUT  512
#define HDIM   56
#define WDIM   56
#define HW     (HDIM * WDIM)
#define NIMG   32
#define NPIX   (NIMG * HW)
#define BN_EPS 1e-5f

__global__ __launch_bounds__(256) void prep_kernel(
    const float* __restrict__ w_dw,
    const float* __restrict__ dw_gamma, const float* __restrict__ dw_beta,
    const float* __restrict__ dw_mean,  const float* __restrict__ dw_var,
    const float* __restrict__ w_pw,
    const float* __restrict__ pw_gamma, const float* __restrict__ pw_beta,
    const float* __restrict__ pw_mean,  const float* __restrict__ pw_var,
    int8_t* __restrict__ signW,
    float*  __restrict__ scale_pw,
    float*  __restrict__ pw_inv, float* __restrict__ pw_bias,
    float*  __restrict__ bwDW,
    float*  __restrict__ dw_inv, float* __restrict__ dw_bias)
{
    int tid = blockIdx.x * blockDim.x + threadIdx.x;
    if (tid < C_OUT) {
        const float* row = w_pw + (size_t)tid * C_IN;
        float s = 0.0f;
        for (int c = 0; c < C_IN; ++c) s += fabsf(row[c]);
        s *= (1.0f / (float)C_IN);
        scale_pw[tid] = s;
        for (int c = 0; c < C_IN; ++c) {
            float w = row[c];
            signW[(size_t)tid * C_IN + c] = (int8_t)((w > 0.0f) - (w < 0.0f));
        }
        float inv = pw_gamma[tid] * rsqrtf(pw_var[tid] + BN_EPS);
        pw_inv[tid]  = inv;
        pw_bias[tid] = pw_beta[tid] - pw_mean[tid] * inv;
    } else if (tid < C_OUT + C_IN) {
        int c = tid - C_OUT;
        const float* row = w_dw + (size_t)c * 9;
        float s = 0.0f;
        for (int j = 0; j < 9; ++j) s += fabsf(row[j]);
        s *= (1.0f / 9.0f);
        for (int j = 0; j < 9; ++j) {
            float w = row[j];
            bwDW[c * 9 + j] = (float)((w > 0.0f) - (w < 0.0f)) * s;
        }
        float inv = dw_gamma[c] * rsqrtf(dw_var[c] + BN_EPS);
        dw_inv[c]  = inv;
        dw_bias[c] = dw_beta[c] - dw_mean[c] * inv;
    }
}

__global__ __launch_bounds__(256) void dw_kernel(
    const float* __restrict__ x,
    const float* __restrict__ bwDW,
    const float* __restrict__ dw_inv, const float* __restrict__ dw_bias,
    uint8_t* __restrict__ actBits)
{
    int pix = blockIdx.x * blockDim.x + threadIdx.x;
    if (pix >= HW) return;
    int c = blockIdx.y;
    int n = blockIdx.z;

    int h = pix / WDIM;
    int w = pix - h * WDIM;

    const float* xc = x + ((size_t)(n * C_IN + c)) * HW;
    const float* wk = bwDW + c * 9;

    float acc = 0.0f;
#pragma unroll
    for (int dy = -1; dy <= 1; ++dy) {
        int hh = h + dy;
        if (hh < 0 || hh >= HDIM) continue;
#pragma unroll
        for (int dx = -1; dx <= 1; ++dx) {
            int ww = w + dx;
            if (ww < 0 || ww >= WDIM) continue;
            acc += xc[hh * WDIM + ww] * wk[(dy + 1) * 3 + (dx + 1)];
        }
    }
    float y = acc * dw_inv[c] + dw_bias[c];
    float t = rintf(y * (1.0f / 6.0f));
    uint8_t bit = (t >= 1.0f) ? 1u : 0u;

    actBits[((size_t)(n * HW + pix)) * C_IN + c] = bit;
}

__device__ __forceinline__ v8i loadA(const int8_t* aRow, int kBase, int half) {
    v8i a;
#pragma unroll
    for (int v = 0; v < 8; ++v) {
        int k = kBase + 32 * (v >> 2) + 16 * ((v >> 1) & 1) + 8 * half + 4 * (v & 1);
        a[v] = *(const int*)(aRow + k);
    }
    return a;
}
__device__ __forceinline__ v8i loadB(const uint8_t* bRow, int kBase, int half) {
    v8i b;
#pragma unroll
    for (int v = 0; v < 8; ++v) {
        int k = kBase + 32 * (v >> 2) + 16 * ((v >> 1) & 1) + 8 * half + 4 * (v & 1);
        b[v] = *(const int*)(bRow + k);
    }
    return b;
}

__global__ __launch_bounds__(128) void pw_wmma_kernel(
    const uint8_t* __restrict__ actBits,
    const int8_t*  __restrict__ signW,
    const float*   __restrict__ scale_pw,
    const float*   __restrict__ pw_inv, const float* __restrict__ pw_bias,
    float* __restrict__ out)
{
    const int lane   = threadIdx.x & 31;
    const int waveId = threadIdx.x >> 5;
    const int waveGl = blockIdx.x * 4 + waveId;
    const int nImg   = blockIdx.y;

    const int oTile    = waveGl & 31;
    const int pixGroup = waveGl >> 5;
    const int oBase    = oTile * 16;
    const int pixBase  = pixGroup * 64;

    const int half = lane >> 4;
    const int l16  = lane & 15;

    const int8_t*  aRow  = signW + (size_t)(oBase + l16) * C_IN;
    const uint8_t* bRow0 = actBits + ((size_t)nImg * HW + pixBase +  0 + l16) * C_IN;
    const uint8_t* bRow1 = bRow0 + 16 * C_IN;
    const uint8_t* bRow2 = bRow0 + 32 * C_IN;
    const uint8_t* bRow3 = bRow0 + 48 * C_IN;

    v8i acc0 = {0,0,0,0,0,0,0,0};
    v8i acc1 = {0,0,0,0,0,0,0,0};
    v8i acc2 = {0,0,0,0,0,0,0,0};
    v8i acc3 = {0,0,0,0,0,0,0,0};

    v8i a  = loadA(aRow, 0, half);
    v8i b0 = loadB(bRow0, 0, half);
    v8i b1 = loadB(bRow1, 0, half);
    v8i b2 = loadB(bRow2, 0, half);
    v8i b3 = loadB(bRow3, 0, half);

#pragma unroll
    for (int kb = 0; kb < 4; ++kb) {
        v8i an, bn0, bn1, bn2, bn3;
        if (kb < 3) {
            const int kN = (kb + 1) * 64;
            an  = loadA(aRow, kN, half);
            bn0 = loadB(bRow0, kN, half);
            bn1 = loadB(bRow1, kN, half);
            bn2 = loadB(bRow2, kN, half);
            bn3 = loadB(bRow3, kN, half);
        }
        __builtin_amdgcn_sched_barrier(0);

        acc0 = __builtin_amdgcn_wmma_i32_16x16x64_iu8(true, a, false, b0, acc0, false, false);
        acc1 = __builtin_amdgcn_wmma_i32_16x16x64_iu8(true, a, false, b1, acc1, false, false);
        acc2 = __builtin_amdgcn_wmma_i32_16x16x64_iu8(true, a, false, b2, acc2, false, false);
        acc3 = __builtin_amdgcn_wmma_i32_16x16x64_iu8(true, a, false, b3, acc3, false, false);

        __builtin_amdgcn_sched_barrier(0);

        if (kb < 3) {
            a = an; b0 = bn0; b1 = bn1; b2 = bn2; b3 = bn3;
        }
    }

    float sc[8], inv[8], bia[8];
#pragma unroll
    for (int r = 0; r < 8; ++r) {
        int o = oBase + r + 8 * half;
        sc[r]  = 6.0f * scale_pw[o];
        inv[r] = pw_inv[o];
        bia[r] = pw_bias[o];
    }

#pragma unroll
    for (int g = 0; g < 4; ++g) {
        v8i accv;
        if      (g == 0) accv = acc0;
        else if (g == 1) accv = acc1;
        else if (g == 2) accv = acc2;
        else             accv = acc3;
        const int p = pixBase + g * 16 + l16;
#pragma unroll
        for (int r = 0; r < 8; ++r) {
            int o = oBase + r + 8 * half;
            float y = sc[r] * (float)accv[r] * inv[r] + bia[r];
            float q = fminf(fmaxf(rintf(y * (1.0f / 6.0f)), 0.0f), 1.0f) * 6.0f;
            out[((size_t)(nImg * C_OUT + o)) * HW + p] = q;
        }
    }
}

extern "C" void kernel_launch(void* const* d_in, const int* in_sizes, int n_in,
                              void* d_out, int out_size, void* d_ws, size_t ws_size,
                              hipStream_t stream) {
    const float* x        = (const float*)d_in[0];
    const float* w_dw     = (const float*)d_in[1];
    const float* dw_gamma = (const float*)d_in[2];
    const float* dw_beta  = (const float*)d_in[3];
    const float* dw_mean  = (const float*)d_in[4];
    const float* dw_var   = (const float*)d_in[5];
    const float* w_pw     = (const float*)d_in[6];
    const float* pw_gamma = (const float*)d_in[7];
    const float* pw_beta  = (const float*)d_in[8];
    const float* pw_mean  = (const float*)d_in[9];
    const float* pw_var   = (const float*)d_in[10];
    float* out = (float*)d_out;

    uint8_t* ws = (uint8_t*)d_ws;
    size_t off = 0;
    auto carve = [&](size_t bytes) -> uint8_t* {
        uint8_t* p = ws + off;
        off = (off + bytes + 255) & ~(size_t)255;
        return p;
    };
    uint8_t* actBits  = carve((size_t)NPIX * C_IN);
    int8_t*  signW    = (int8_t*)carve((size_t)C_OUT * C_IN);
    float*   bwDW     = (float*)carve(C_IN * 9 * sizeof(float));
    float*   scale_pw = (float*)carve(C_OUT * sizeof(float));
    float*   pw_inv   = (float*)carve(C_OUT * sizeof(float));
    float*   pw_bias  = (float*)carve(C_OUT * sizeof(float));
    float*   dw_inv   = (float*)carve(C_IN * sizeof(float));
    float*   dw_bias  = (float*)carve(C_IN * sizeof(float));
    (void)ws_size; (void)in_sizes; (void)n_in; (void)out_size;

    prep_kernel<<<3, 256, 0, stream>>>(
        w_dw, dw_gamma, dw_beta, dw_mean, dw_var,
        w_pw, pw_gamma, pw_beta, pw_mean, pw_var,
        signW, scale_pw, pw_inv, pw_bias, bwDW, dw_inv, dw_bias);

    {
        dim3 grid((HW + 255) / 256, C_IN, NIMG);
        dw_kernel<<<grid, 256, 0, stream>>>(x, bwDW, dw_inv, dw_bias, actBits);
    }

    {
        dim3 grid((32 * 49) / 4, NIMG);
        pw_wmma_kernel<<<grid, 128, 0, stream>>>(
            actBits, signW, scale_pw, pw_inv, pw_bias, out);
    }
}
